// ContrastiveLoss_29953101922499
// MI455X (gfx1250) — compile-verified, matrix operand order checked by reading, not run on hardware
//
#include <hip/hip_runtime.h>
#include <hip/hip_bf16.h>
#include <math.h>

typedef __attribute__((ext_vector_type(16))) __bf16 v16bf;
typedef __attribute__((ext_vector_type(8)))  float  v8f;
typedef __attribute__((ext_vector_type(4)))  int    v4i;

typedef __attribute__((address_space(1))) v4i gv4i;   // global int4
typedef __attribute__((address_space(3))) v4i lv4i;   // LDS int4

#define BATCH 8192
#define DIM   256
#define INV_T 2.0f   // 1 / 0.5
#define TILE_N 64    // B-tile rows (columns of the sim matrix) per stage

#if defined(__has_builtin)
#if __has_builtin(__builtin_amdgcn_global_load_async_to_lds_b128)
#define HAVE_ASYNC_LDS 1
#endif
#endif

__device__ __forceinline__ void wait_asynccnt0() {
#if defined(__has_builtin) && __has_builtin(__builtin_amdgcn_s_wait_asynccnt)
    __builtin_amdgcn_s_wait_asynccnt(0);
#else
    asm volatile("s_wait_asynccnt 0x0" ::: "memory");
#endif
}

__device__ __forceinline__ unsigned short f2bf(float f) {
    unsigned u = __float_as_uint(f);
    u += 0x7FFFu + ((u >> 16) & 1u);   // round-to-nearest-even
    return (unsigned short)(u >> 16);
}

// ---------------------------------------------------------------------------
// Kernel 1: L2-normalize rows, emit bf16 z_i / z_k, pos = diag(z_i z_j^T),
// and zero rowsum (deterministic re-init every call).
// One wave (32 lanes) per row; 8 floats per lane.
// ---------------------------------------------------------------------------
__global__ __launch_bounds__(256) void norm_kernel(
    const float* __restrict__ ei, const float* __restrict__ ej,
    const float* __restrict__ ek,
    unsigned short* __restrict__ zi, unsigned short* __restrict__ zk,
    float* __restrict__ pos, float* __restrict__ rowsum)
{
    const int wave = threadIdx.x >> 5;
    const int lane = threadIdx.x & 31;
    const int row  = blockIdx.x * 8 + wave;
    const size_t base = (size_t)row * DIM + lane * 8;

    __align__(16) float fi[8], fj[8], fk[8];
    *(float4*)(fi)     = ((const float4*)(ei + base))[0];
    *(float4*)(fi + 4) = ((const float4*)(ei + base))[1];
    *(float4*)(fj)     = ((const float4*)(ej + base))[0];
    *(float4*)(fj + 4) = ((const float4*)(ej + base))[1];
    *(float4*)(fk)     = ((const float4*)(ek + base))[0];
    *(float4*)(fk + 4) = ((const float4*)(ek + base))[1];

    float ssi = 0.f, ssj = 0.f, ssk = 0.f, dij = 0.f;
#pragma unroll
    for (int t = 0; t < 8; ++t) {
        ssi += fi[t] * fi[t];
        ssj += fj[t] * fj[t];
        ssk += fk[t] * fk[t];
        dij += fi[t] * fj[t];
    }
    // full wave32 xor reduction (every lane ends with the totals)
#pragma unroll
    for (int m = 16; m >= 1; m >>= 1) {
        ssi += __shfl_xor(ssi, m, 32);
        ssj += __shfl_xor(ssj, m, 32);
        ssk += __shfl_xor(ssk, m, 32);
        dij += __shfl_xor(dij, m, 32);
    }
    const float inv_i = 1.0f / fmaxf(sqrtf(ssi), 1e-12f);
    const float inv_j = 1.0f / fmaxf(sqrtf(ssj), 1e-12f);
    const float inv_k = 1.0f / fmaxf(sqrtf(ssk), 1e-12f);

    if (lane == 0) pos[row]    = dij * inv_i * inv_j;
    if (lane == 1) rowsum[row] = 0.0f;

    __align__(16) unsigned short oi[8], ok[8];
#pragma unroll
    for (int t = 0; t < 8; ++t) {
        oi[t] = f2bf(fi[t] * inv_i);
        ok[t] = f2bf(fk[t] * inv_k);
    }
    *(uint4*)(zi + base) = *(uint4*)oi;
    *(uint4*)(zk + base) = *(uint4*)ok;
}

// ---------------------------------------------------------------------------
// Kernel 2: fused  rowsum[m] += sum_n exp( (z_i[m] . z_k[n]) / T )
// Block = 8 waves; wave owns a 16-row A strip (all K=256 in registers).
// Double-buffered 64x256 bf16 z_k tiles in LDS, filled with async
// global->LDS copies (ASYNCcnt) when available, overlapping the
// v_wmma_f32_16x16x32_bf16 inner loop; one barrier per tile.
// Grid.y slices the 8192-column N range 8 ways (atomic float accumulate).
// ---------------------------------------------------------------------------
__global__ __launch_bounds__(256) void gemm_rowsum_kernel(
    const unsigned short* __restrict__ zi,
    const unsigned short* __restrict__ zk,
    float* __restrict__ rowsum)
{
    __shared__ __align__(16) unsigned short sB[2 * TILE_N * DIM];  // 2 x 32 KB

    const int wave = threadIdx.x >> 5;
    const int lane = threadIdx.x & 31;
    const int c    = lane >> 4;     // lane half-group (K-chunk selector)
    const int mr   = lane & 15;     // row (A) / column-row (B) within tile
    const int m0   = blockIdx.x * 128 + wave * 16;

    // Load A fragments: 16x256 bf16 strip, ISA 16-bit A layout.
    // Lane group c holds K = kc*32 + c*8 + [0..7]  and  kc*32 + 16 + c*8 + [0..7]
    v16bf A[8];
    {
        const unsigned short* rowA = zi + (size_t)(m0 + mr) * DIM;
#pragma unroll
        for (int kc = 0; kc < 8; ++kc) {
            ((uint4*)&A[kc])[0] = *(const uint4*)(rowA + kc * 32 + c * 8);
            ((uint4*)&A[kc])[1] = *(const uint4*)(rowA + kc * 32 + 16 + c * 8);
        }
    }

    const int n_base  = blockIdx.y * 1024;
    const int n_tiles = 1024 / TILE_N;   // 16

#if HAVE_ASYNC_LDS
    // ---- async double-buffered pipeline -----------------------------------
    // issue: 2048 uint4 per tile -> 8 x b128 async transfers per thread
    auto issue_tile = [&](int t) {
        const uint4* src = (const uint4*)(zk + (size_t)(n_base + t * TILE_N) * DIM);
        uint4*       dst = (uint4*)(sB + (size_t)(t & 1) * TILE_N * DIM);
#pragma unroll
        for (int i = 0; i < (TILE_N * DIM / 8) / 256; ++i) {
            const int idx = threadIdx.x + i * 256;
            __builtin_amdgcn_global_load_async_to_lds_b128(
                (gv4i*)(src + idx), (lv4i*)(dst + idx), 0, 0);
        }
    };
    issue_tile(0);
    wait_asynccnt0();
    __syncthreads();
#endif

    v8f rs = {};                       // per-lane row partial sums
    for (int t = 0; t < n_tiles; ++t) {
#if HAVE_ASYNC_LDS
        if (t + 1 < n_tiles) issue_tile(t + 1);    // overlap next fill
        const unsigned short* buf = sB + (size_t)(t & 1) * TILE_N * DIM;
#else
        // ---- synchronous staging fallback ---------------------------------
        __syncthreads();
        {
            const uint4* src = (const uint4*)(zk + (size_t)(n_base + t * TILE_N) * DIM);
            uint4*       dst = (uint4*)sB;
#pragma unroll
            for (int i = 0; i < (TILE_N * DIM / 8) / 256; ++i)
                dst[threadIdx.x + i * 256] = src[threadIdx.x + i * 256];
        }
        __syncthreads();
        const unsigned short* buf = sB;
#endif

#pragma unroll
        for (int ns = 0; ns < TILE_N / 16; ++ns) {
            const unsigned short* rowB = buf + (size_t)(ns * 16 + mr) * DIM;
            v8f acc = {};
#pragma unroll
            for (int kc = 0; kc < 8; ++kc) {
                v16bf Bf;
                ((uint4*)&Bf)[0] = *(const uint4*)(rowB + kc * 32 + c * 8);
                ((uint4*)&Bf)[1] = *(const uint4*)(rowB + kc * 32 + 16 + c * 8);
                acc = __builtin_amdgcn_wmma_f32_16x16x32_bf16(
                    false, A[kc], false, Bf, (short)0, acc, false, false);
            }
            // C layout: VGPR r, lane l -> M = r + 8*(l/16), N = l%16
#pragma unroll
            for (int r = 0; r < 8; ++r)
                rs[r] += __expf(acc[r] * INV_T);
        }

#if HAVE_ASYNC_LDS
        wait_asynccnt0();   // own async transfers for t+1 landed
        __syncthreads();    // everyone's transfers landed; compute-t done
#endif
    }

    // reduce over the 16 lanes of each half (columns of the tiles)
#pragma unroll
    for (int m = 1; m < 16; m <<= 1) {
#pragma unroll
        for (int r = 0; r < 8; ++r) {
            float v = rs[r];
            rs[r] = v + __shfl_xor(v, m, 16);
        }
    }
    if (mr == 0) {
#pragma unroll
        for (int r = 0; r < 8; ++r)
            atomicAdd(&rowsum[m0 + c * 8 + r], rs[r]);
    }
}

// ---------------------------------------------------------------------------
// Kernel 3: loss = mean( log(exp(pos/T) + rowsum) - pos/T )
// ---------------------------------------------------------------------------
__global__ __launch_bounds__(256) void loss_kernel(
    const float* __restrict__ pos, const float* __restrict__ rowsum,
    float* __restrict__ out)
{
    __shared__ float red[8];
    float part = 0.f;
    for (int i = threadIdx.x; i < BATCH; i += 256) {
        const float p = pos[i] * INV_T;
        part += __logf(__expf(p) + rowsum[i]) - p;
    }
#pragma unroll
    for (int m = 16; m >= 1; m >>= 1)
        part += __shfl_xor(part, m, 32);
    const int wave = threadIdx.x >> 5;
    if ((threadIdx.x & 31) == 0) red[wave] = part;
    __syncthreads();
    if (threadIdx.x == 0) {
        float tot = 0.f;
#pragma unroll
        for (int w = 0; w < 8; ++w) tot += red[w];
        out[0] = tot / (float)BATCH;
    }
}

// ---------------------------------------------------------------------------
extern "C" void kernel_launch(void* const* d_in, const int* in_sizes, int n_in,
                              void* d_out, int out_size, void* d_ws, size_t ws_size,
                              hipStream_t stream) {
    const float* ei = (const float*)d_in[0];
    const float* ej = (const float*)d_in[1];
    const float* ek = (const float*)d_in[2];

    char* ws = (char*)d_ws;
    unsigned short* zi = (unsigned short*)ws;                              // 4 MiB
    unsigned short* zk = (unsigned short*)(ws + (size_t)BATCH * DIM * 2);  // 4 MiB
    float* pos    = (float*)(ws + 2 * (size_t)BATCH * DIM * 2);            // 32 KiB
    float* rowsum = pos + BATCH;                                           // 32 KiB

    norm_kernel<<<BATCH / 8, 256, 0, stream>>>(ei, ej, ek, zi, zk, pos, rowsum);

    dim3 g2(BATCH / 128, 8);
    gemm_rowsum_kernel<<<g2, 256, 0, stream>>>(zi, zk, rowsum);

    loss_kernel<<<1, 256, 0, stream>>>(pos, rowsum, (float*)d_out);
}
